// AutoRegressiveTransformer_63625645523436
// MI455X (gfx1250) — compile-verified, matrix operand order checked by reading, not run on hardware
//
#include <hip/hip_runtime.h>
#include <hip/hip_bf16.h>
#include <math.h>

// ---------------- problem constants (from reference) ----------------
#define BB   4
#define TT1  768
#define TT2  256
#define TT   1024          // TT1 + TT2
#define DD   1024
#define HM   4096          // MLP hidden
#define LL   12
#define VV   512
#define MM   (BB*TT)       // 4096 flattened tokens

#define LDSK 40            // padded LDS row stride in bf16 (32 used + 8 pad = 80B, 16B-aligned)

typedef __bf16 bf16_t;
typedef __attribute__((ext_vector_type(16))) __bf16 v16bf;
typedef __attribute__((ext_vector_type(8)))  __bf16 v8bf;
typedef __attribute__((ext_vector_type(4)))  __bf16 v4bf;
typedef __attribute__((ext_vector_type(8)))  float  v8f;

enum { EPI_STORE = 0, EPI_BIAS = 1, EPI_BIAS_GELU = 2, EPI_BIAS_ADD = 3, EPI_ADD_SCALED = 4 };

// ---------------- embedding: H0 = concat(i_emb[zi], t_emb[xt]) + pos ----------------
__global__ __launch_bounds__(256)
void embed_kernel(const int* __restrict__ xt, const int* __restrict__ zi,
                  const float* __restrict__ pos, const float* __restrict__ temb,
                  const float* __restrict__ iemb, float* __restrict__ H)
{
    long q = (long)blockIdx.x * 256 + threadIdx.x;        // over MM*DD/4 float4s
    int d4 = (int)(q & (DD / 4 - 1)) * 4;
    int t  = (int)((q >> 8) & (TT - 1));
    int b  = (int)(q >> 18);
    int tok; const float* emb;
    if (t < TT2) { tok = zi[b * TT2 + t];         emb = iemb; }
    else         { tok = xt[b * TT1 + (t - TT2)]; emb = temb; }
    const float4 e = *(const float4*)&emb[(long)tok * DD + d4];
    const float4 p = *(const float4*)&pos[(long)t * DD + d4];
    float4 o; o.x = e.x + p.x; o.y = e.y + p.y; o.z = e.z + p.z; o.w = e.w + p.w;
    *(float4*)&H[q * 4] = o;
}

// ---------------- layernorm: one block per row of length DD ----------------
__global__ __launch_bounds__(256)
void ln_kernel(const float* __restrict__ X, const float* __restrict__ g,
               const float* __restrict__ b, float* __restrict__ Y)
{
    __shared__ float s1[256];
    __shared__ float s2[256];
    const int tid = threadIdx.x;
    const float* x = X + (long)blockIdx.x * DD;
    const float4 v = *(const float4*)&x[tid * 4];
    float a = v.x + v.y + v.z + v.w;
    float q = v.x * v.x + v.y * v.y + v.z * v.z + v.w * v.w;
    s1[tid] = a; s2[tid] = q;
    __syncthreads();
    for (int off = 128; off > 0; off >>= 1) {
        if (tid < off) { s1[tid] += s1[tid + off]; s2[tid] += s2[tid + off]; }
        __syncthreads();
    }
    float mean = s1[0] * (1.0f / DD);
    float var  = s2[0] * (1.0f / DD) - mean * mean;
    float rstd = rsqrtf(var + 1e-5f);
    const float4 gg = *(const float4*)&g[tid * 4];
    const float4 bb = *(const float4*)&b[tid * 4];
    float4 o;
    o.x = (v.x - mean) * rstd * gg.x + bb.x;
    o.y = (v.y - mean) * rstd * gg.y + bb.y;
    o.z = (v.z - mean) * rstd * gg.z + bb.z;
    o.w = (v.w - mean) * rstd * gg.w + bb.w;
    *(float4*)&(Y + (long)blockIdx.x * DD)[tid * 4] = o;
}

// ---------------- row softmax over TT with pre-scale ----------------
__global__ __launch_bounds__(256)
void softmax_kernel(float* __restrict__ S, float scale)
{
    __shared__ float sh[256];
    const int tid = threadIdx.x;
    float* x = S + (long)blockIdx.x * TT;
    float4 v = *(const float4*)&x[tid * 4];
    v.x *= scale; v.y *= scale; v.z *= scale; v.w *= scale;
    float mx = fmaxf(fmaxf(v.x, v.y), fmaxf(v.z, v.w));
    sh[tid] = mx; __syncthreads();
    for (int off = 128; off > 0; off >>= 1) {
        if (tid < off) sh[tid] = fmaxf(sh[tid], sh[tid + off]);
        __syncthreads();
    }
    mx = sh[0];
    __syncthreads();
    v.x = expf(v.x - mx); v.y = expf(v.y - mx); v.z = expf(v.z - mx); v.w = expf(v.w - mx);
    sh[tid] = v.x + v.y + v.z + v.w; __syncthreads();
    for (int off = 128; off > 0; off >>= 1) {
        if (tid < off) sh[tid] += sh[tid + off];
        __syncthreads();
    }
    float inv = 1.0f / sh[0];
    v.x *= inv; v.y *= inv; v.z *= inv; v.w *= inv;
    *(float4*)&x[tid * 4] = v;
}

// ---------------- WMMA bf16 GEMM ----------------
// C[M,N] = epilogue( A[M,K] @ (BT ? B[N,K]^T : B[K,N]) ), batched via blockIdx.z.
// 128x128 block tile, 8 wave32 waves (4x2), each wave 2x4 grid of 16x16x32 WMMAs.
// f32 operands converted to bf16 while staged into double-buffered, row-padded LDS;
// register-prefetched pipeline with one barrier per K-step.
template <bool BT, int EPI>
__global__ __launch_bounds__(256)
void gemm_wmma(const float* __restrict__ A, const float* __restrict__ Bm,
               float* __restrict__ C, const float* __restrict__ bias,
               int Kdim, int lda, int ldb, int ldc,
               long strideA, long strideB, long strideC, float scale)
{
    __shared__ bf16_t As[2][128 * LDSK];   // [m][k], 80B padded rows
    __shared__ bf16_t Bs[2][128 * LDSK];   // [n][k], 80B padded rows

    const int tid  = threadIdx.x;
    const int lane = tid & 31;
    const int wave = tid >> 5;
    const int wm0  = (wave & 3) * 32;   // 4 waves along M
    const int wn0  = (wave >> 2) * 64;  // 2 waves along N
    const int hf   = lane >> 4;         // lane half (ISA 16-bit operand layout)
    const int l16  = lane & 15;

    const int row0 = blockIdx.y * 128;
    const int col0 = blockIdx.x * 128;

    const float* Ab = A  + (long)blockIdx.z * strideA;
    const float* Bb = Bm + (long)blockIdx.z * strideB;
    float*       Cb = C  + (long)blockIdx.z * strideC;

    // compile-time staging decomposition (straight-line b128 loads, no masked loops)
    const int am  = tid >> 3;           // A/BT-B: base row, +32 per iteration
    const int ak  = (tid & 7) * 4;      // A/BT-B: 16B-aligned k offset
    const int bk4 = (tid >> 5) * 4;     // B[K,N]: 4-row k block per wave
    const int bn  = (tid & 31) * 4;     // B[K,N]: 16B-aligned n offset

    float4 aR[4], bR[4];

    auto loadA = [&](int kt) {
#pragma unroll
        for (int i = 0; i < 4; ++i)
            aR[i] = *(const float4*)&Ab[(long)(row0 + am + 32 * i) * lda + kt + ak];
    };
    auto loadB = [&](int kt) {
        if (BT) {
#pragma unroll
            for (int i = 0; i < 4; ++i)
                bR[i] = *(const float4*)&Bb[(long)(col0 + am + 32 * i) * ldb + kt + ak];
        } else {
            // 4x4 sub-block: rows kt+bk4..+3, cols col0+bn..+3 (coalesced per wave)
#pragma unroll
            for (int i = 0; i < 4; ++i)
                bR[i] = *(const float4*)&Bb[(long)(kt + bk4 + i) * ldb + col0 + bn];
            // stream hint for the weight matrix two K-tiles ahead -> global_prefetch
            if (kt + 64 < Kdim)
                __builtin_prefetch(&Bb[(long)(kt + 64 + bk4) * ldb + col0 + bn], 0, 1);
        }
    };
    auto storeA = [&](int p) {
#pragma unroll
        for (int i = 0; i < 4; ++i) {
            v4bf w;
            w[0] = (bf16_t)aR[i].x; w[1] = (bf16_t)aR[i].y;
            w[2] = (bf16_t)aR[i].z; w[3] = (bf16_t)aR[i].w;
            *(v4bf*)&As[p][(am + 32 * i) * LDSK + ak] = w;
        }
    };
    auto storeB = [&](int p) {
        if (BT) {
#pragma unroll
            for (int i = 0; i < 4; ++i) {
                v4bf w;
                w[0] = (bf16_t)bR[i].x; w[1] = (bf16_t)bR[i].y;
                w[2] = (bf16_t)bR[i].z; w[3] = (bf16_t)bR[i].w;
                *(v4bf*)&Bs[p][(am + 32 * i) * LDSK + ak] = w;
            }
        } else {
            // register 4x4 transpose during the bf16 pack: [k][n] -> [n][k],
            // 4x 8-byte stores instead of 16x b16 scatters
            v4bf w0, w1, w2, w3;
            w0[0] = (bf16_t)bR[0].x; w0[1] = (bf16_t)bR[1].x; w0[2] = (bf16_t)bR[2].x; w0[3] = (bf16_t)bR[3].x;
            w1[0] = (bf16_t)bR[0].y; w1[1] = (bf16_t)bR[1].y; w1[2] = (bf16_t)bR[2].y; w1[3] = (bf16_t)bR[3].y;
            w2[0] = (bf16_t)bR[0].z; w2[1] = (bf16_t)bR[1].z; w2[2] = (bf16_t)bR[2].z; w2[3] = (bf16_t)bR[3].z;
            w3[0] = (bf16_t)bR[0].w; w3[1] = (bf16_t)bR[1].w; w3[2] = (bf16_t)bR[2].w; w3[3] = (bf16_t)bR[3].w;
            *(v4bf*)&Bs[p][(bn + 0) * LDSK + bk4] = w0;
            *(v4bf*)&Bs[p][(bn + 1) * LDSK + bk4] = w1;
            *(v4bf*)&Bs[p][(bn + 2) * LDSK + bk4] = w2;
            *(v4bf*)&Bs[p][(bn + 3) * LDSK + bk4] = w3;
        }
    };

    v8f acc[2][4];
#pragma unroll
    for (int i = 0; i < 2; ++i)
#pragma unroll
        for (int j = 0; j < 4; ++j)
            acc[i][j] = (v8f){0.f, 0.f, 0.f, 0.f, 0.f, 0.f, 0.f, 0.f};

    const int nk = Kdim >> 5;
    loadA(0); loadB(0);
    storeA(0); storeB(0);
    __syncthreads();

    for (int it = 0; it < nk; ++it) {
        const int p = it & 1;
        const bool more = (it + 1) < nk;
        if (more) { loadA((it + 1) << 5); loadB((it + 1) << 5); }

        // Fragments per ISA 16-bit A (16x32) / B (32x16) layouts:
        // lane half 0 holds K 0..7 & 16..23, half 1 holds K 8..15 & 24..31,
        // i.e. two contiguous 8xbf16 (16B) runs per fragment -> ds_load_b128.
        v16bf aF[2], bF[4];
#pragma unroll
        for (int im = 0; im < 2; ++im) {
            int m = wm0 + im * 16 + l16;
            v8bf lo = *(const v8bf*)&As[p][m * LDSK + 8 * hf];
            v8bf hi = *(const v8bf*)&As[p][m * LDSK + 16 + 8 * hf];
#pragma unroll
            for (int e = 0; e < 8; ++e) { aF[im][e] = lo[e]; aF[im][e + 8] = hi[e]; }
        }
#pragma unroll
        for (int jn = 0; jn < 4; ++jn) {
            int n = wn0 + jn * 16 + l16;
            v8bf lo = *(const v8bf*)&Bs[p][n * LDSK + 8 * hf];
            v8bf hi = *(const v8bf*)&Bs[p][n * LDSK + 16 + 8 * hf];
#pragma unroll
            for (int e = 0; e < 8; ++e) { bF[jn][e] = lo[e]; bF[jn][e + 8] = hi[e]; }
        }
#pragma unroll
        for (int im = 0; im < 2; ++im)
#pragma unroll
            for (int jn = 0; jn < 4; ++jn)
                acc[im][jn] = __builtin_amdgcn_wmma_f32_16x16x32_bf16(
                    false, aF[im], false, bF[jn], (short)0, acc[im][jn], false, false);

        if (more) {
            storeA(p ^ 1); storeB(p ^ 1);   // safe: buffer p^1 last read before prev barrier
            __syncthreads();
        }
    }

    // Epilogue. f32 C/D layout: VGPR r -> row r (lanes 0-15) / row r+8 (lanes 16-31).
#pragma unroll
    for (int im = 0; im < 2; ++im) {
#pragma unroll
        for (int jn = 0; jn < 4; ++jn) {
            int gcol = col0 + wn0 + jn * 16 + l16;
#pragma unroll
            for (int r = 0; r < 8; ++r) {
                int grow = row0 + wm0 + im * 16 + r + 8 * hf;
                long idx = (long)grow * ldc + gcol;
                float v = acc[im][jn][r];
                if constexpr (EPI == EPI_STORE) {
                    Cb[idx] = v;
                } else if constexpr (EPI == EPI_BIAS) {
                    Cb[idx] = v + bias[gcol];
                } else if constexpr (EPI == EPI_BIAS_GELU) {
                    float x = v + bias[gcol];
                    Cb[idx] = 0.5f * x * (1.0f + erff(x * 0.70710678118654752f));
                } else if constexpr (EPI == EPI_BIAS_ADD) {
                    Cb[idx] = Cb[idx] + v + bias[gcol];
                } else { // EPI_ADD_SCALED: residual += scale * acc
                    Cb[idx] = Cb[idx] + scale * v;
                }
            }
        }
    }
}

// ---------------- driver ----------------
extern "C" void kernel_launch(void* const* d_in, const int* in_sizes, int n_in,
                              void* d_out, int out_size, void* d_ws, size_t ws_size,
                              hipStream_t stream)
{
    (void)in_sizes; (void)n_in; (void)out_size; (void)ws_size;
    const int*   xt   = (const int*)  d_in[0];
    const int*   zi   = (const int*)  d_in[1];
    const float* pos  = (const float*)d_in[2];
    const float* temb = (const float*)d_in[3];
    const float* iemb = (const float*)d_in[4];
    const float* ln1g = (const float*)d_in[5];
    const float* ln1b = (const float*)d_in[6];
    const float* Wq   = (const float*)d_in[7];
    const float* Wk   = (const float*)d_in[8];
    const float* Wv   = (const float*)d_in[9];
    const float* ln2g = (const float*)d_in[10];
    const float* ln2b = (const float*)d_in[11];
    const float* W1   = (const float*)d_in[12];
    const float* b1   = (const float*)d_in[13];
    const float* W2   = (const float*)d_in[14];
    const float* b2   = (const float*)d_in[15];
    const float* roW  = (const float*)d_in[16];
    const float* rob  = (const float*)d_in[17];
    float* out = (float*)d_out;

    // workspace carve-up (~160 MB)
    char* ws = (char*)d_ws;
    float* H   = (float*)ws; ws += (size_t)MM * DD * 4;   // residual stream
    float* H1  = (float*)ws; ws += (size_t)MM * DD * 4;   // LN output
    float* Qb  = (float*)ws; ws += (size_t)MM * DD * 4;
    float* Kb  = (float*)ws; ws += (size_t)MM * DD * 4;
    float* Vb  = (float*)ws; ws += (size_t)MM * DD * 4;
    float* S   = (float*)ws; ws += (size_t)BB * TT * TT * 4;
    float* MID = (float*)ws; ws += (size_t)MM * HM * 4;

    embed_kernel<<<(MM * DD / 4) / 256, 256, 0, stream>>>(xt, zi, pos, temb, iemb, H);

    const float inv_sqrt_d = 0.03125f;              // 1/sqrt(1024)
    const float attn_scale = 1.0f + 1.0f / DD;      // attn@v + (attn/D)@v folded

    for (int l = 0; l < LL; ++l) {
        // --- attention block ---
        ln_kernel<<<MM, 256, 0, stream>>>(H, ln1g + l * DD, ln1b + l * DD, H1);

        dim3 gqkv(DD / 128, MM / 128, 1);
        gemm_wmma<false, EPI_STORE><<<gqkv, 256, 0, stream>>>(
            H1, Wq + (long)l * DD * DD, Qb, nullptr, DD, DD, DD, DD, 0, 0, 0, 1.f);
        gemm_wmma<false, EPI_STORE><<<gqkv, 256, 0, stream>>>(
            H1, Wk + (long)l * DD * DD, Kb, nullptr, DD, DD, DD, DD, 0, 0, 0, 1.f);
        gemm_wmma<false, EPI_STORE><<<gqkv, 256, 0, stream>>>(
            H1, Wv + (long)l * DD * DD, Vb, nullptr, DD, DD, DD, DD, 0, 0, 0, 1.f);

        // scores: S[b] = Q[b] @ K[b]^T   (batched over blockIdx.z)
        dim3 gs(TT / 128, TT / 128, BB);
        gemm_wmma<true, EPI_STORE><<<gs, 256, 0, stream>>>(
            Qb, Kb, S, nullptr, DD, DD, DD, TT,
            (long)TT * DD, (long)TT * DD, (long)TT * TT, 1.f);

        softmax_kernel<<<BB * TT, 256, 0, stream>>>(S, inv_sqrt_d);

        // H += (1 + 1/D) * (attn @ V)
        dim3 ga(DD / 128, TT / 128, BB);
        gemm_wmma<false, EPI_ADD_SCALED><<<ga, 256, 0, stream>>>(
            S, Vb, H, nullptr, TT, TT, DD, DD,
            (long)TT * TT, (long)TT * DD, (long)TT * DD, attn_scale);

        // --- MLP block ---
        ln_kernel<<<MM, 256, 0, stream>>>(H, ln2g + l * DD, ln2b + l * DD, H1);

        dim3 gm1(HM / 128, MM / 128, 1);
        gemm_wmma<false, EPI_BIAS_GELU><<<gm1, 256, 0, stream>>>(
            H1, W1 + (long)l * DD * HM, MID, b1 + (long)l * HM,
            DD, DD, HM, HM, 0, 0, 0, 1.f);

        dim3 gm2(DD / 128, MM / 128, 1);
        gemm_wmma<false, EPI_BIAS_ADD><<<gm2, 256, 0, stream>>>(
            MID, W2 + (long)l * HM * DD, H, b2 + (long)l * DD,
            HM, HM, DD, DD, 0, 0, 0, 1.f);
    }

    // head on the [:, NI:, :] slice only: per batch, rows b*TT+TT2 .. b*TT+TT-1
    dim3 gh(VV / 128, TT1 / 128, BB);
    gemm_wmma<false, EPI_BIAS><<<gh, 256, 0, stream>>>(
        H + (long)TT2 * DD, roW, out, rob, DD, DD, VV, VV,
        (long)TT * DD, 0, (long)TT1 * VV, 1.f);
}
